// Attention_33767032881664
// MI455X (gfx1250) — compile-verified, matrix operand order checked by reading, not run on hardware
//
#include <hip/hip_runtime.h>
#include <hip/hip_bf16.h>

// ---------------- types / constants ----------------
typedef __bf16 bf16_t;
typedef bf16_t v16bf __attribute__((ext_vector_type(16)));
typedef float  v8f   __attribute__((ext_vector_type(8)));

#define BB   2
#define SS   2048
#define DD   2048
#define HH   16
#define KVH  8
#define YLL  256
#define YDD  1024
#define HDD  128
#define NREP (HH / KVH)

__device__ __forceinline__ bf16_t f2bf(float f) {
  unsigned u = __builtin_bit_cast(unsigned, f);
  u += 0x7FFFu + ((u >> 16) & 1u);              // round-to-nearest-even
  unsigned short s = (unsigned short)(u >> 16);
  return __builtin_bit_cast(bf16_t, s);
}

// Per ISA 7.12.2, a 16-bit fragment's 16 elements per lane are two contiguous
// 8-element runs: k = 8*half + e (e<8) and k = 16 + 8*half + (e-8) (e>=8).
struct U32x2 { uint4 a, b; };
__device__ __forceinline__ v16bf load_frag(const bf16_t* __restrict__ run0,
                                           const bf16_t* __restrict__ run1) {
  U32x2 t;
  t.a = *reinterpret_cast<const uint4*>(run0);
  t.b = *reinterpret_cast<const uint4*>(run1);
  return __builtin_bit_cast(v16bf, t);
}

// CDNA5 async DMA: copy 32 contiguous bytes global -> LDS without touching VGPRs.
// IOFFSET is added to BOTH the global and LDS addresses (ISA 10.7/15.18), so the
// second 16B chunk reuses the same operands with offset:16. Tracked by ASYNCcnt.
__device__ __forceinline__ void async_copy_32B(const bf16_t* gsrc, bf16_t* ldst) {
  unsigned l = (unsigned)(unsigned long long)ldst;   // low 32 bits = LDS address
  unsigned long long g = (unsigned long long)gsrc;
  asm volatile("global_load_async_to_lds_b128 %0, %1, off"
               :: "v"(l), "v"(g) : "memory");
  asm volatile("global_load_async_to_lds_b128 %0, %1, off offset:16"
               :: "v"(l), "v"(g) : "memory");
}
__device__ __forceinline__ void wait_async_zero() {
  asm volatile("s_wait_asynccnt 0x0" ::: "memory");
}

// ---------------- fp32 -> bf16 convert ----------------
__global__ void cvt_bf16_kernel(const float* __restrict__ src,
                                bf16_t* __restrict__ dst, long n) {
  long i = (long)blockIdx.x * blockDim.x + threadIdx.x;
  long stride = (long)gridDim.x * blockDim.x;
  for (; i < n; i += stride) dst[i] = f2bf(src[i]);
}

// ---------------- bf16 GEMM: C(f32, MxN) = A(bf16, MxK) * B(bf16, KxN), row-major
// M % 128 == 0, N % 64 == 0, K % 32 == 0. Block = 256 thr = 8 waves.
// A tile staged by async DMA; B tile staged transposed so fragments are contiguous.
__global__ void __launch_bounds__(256)
gemm_bf16_kernel(const bf16_t* __restrict__ A, const bf16_t* __restrict__ B,
                 float* __restrict__ C, int M, int N, int K) {
  __shared__ __attribute__((aligned(16))) bf16_t As[128][40];  // [m][k], padded
  __shared__ __attribute__((aligned(16))) bf16_t Bt[64][40];   // [n][k], padded
  const int tid  = threadIdx.x;
  const int w    = tid >> 5, lane = tid & 31;
  const int half = lane >> 4, lm = lane & 15;
  const int m0 = blockIdx.y * 128, n0 = blockIdx.x * 64;
  const int h8 = half << 3;

  const int ar = tid >> 1, ac = (tid & 1) << 4;     // A staging coords
  const int br = tid >> 3, bc = (tid & 7) << 3;     // B staging coords

  v8f zero = {0.f,0.f,0.f,0.f,0.f,0.f,0.f,0.f};
  v8f acc[4];
#pragma unroll
  for (int i = 0; i < 4; i++) acc[i] = zero;

  for (int k0 = 0; k0 < K; k0 += 32) {
    // stage A (128x32) via async DMA, 32B per thread
    async_copy_32B(A + (long)(m0 + ar) * K + k0 + ac, &As[ar][ac]);
    { // stage B transposed: read 8 contiguous cols of one k-row, scatter to Bt[n][k]
      bf16_t tmp[8];
      *reinterpret_cast<uint4*>(tmp) =
          *reinterpret_cast<const uint4*>(B + (long)(k0 + br) * N + n0 + bc);
#pragma unroll
      for (int i = 0; i < 8; i++) Bt[bc + i][br] = tmp[i];
    }
    if (k0 + 32 < K) { // prefetch next k-step tiles
      __builtin_prefetch(A + (long)(m0 + ar) * K + k0 + 32 + ac, 0, 0);
      __builtin_prefetch(B + (long)(k0 + 32 + br) * N + n0 + bc, 0, 0);
    }
    wait_async_zero();
    __syncthreads();

    const int am = (w << 4) + lm;
    v16bf af = load_frag(&As[am][h8], &As[am][16 + h8]);
#pragma unroll
    for (int nt = 0; nt < 4; nt++) {
      const int bn = (nt << 4) + lm;
      v16bf bv = load_frag(&Bt[bn][h8], &Bt[bn][16 + h8]);
      acc[nt] = __builtin_amdgcn_wmma_f32_16x16x32_bf16(
          false, af, false, bv, (short)0, acc[nt], false, false);
    }
    __syncthreads();
  }

#pragma unroll
  for (int nt = 0; nt < 4; nt++)
#pragma unroll
    for (int r = 0; r < 8; r++)
      C[(long)(m0 + (w << 4) + r + (half << 3)) * N + n0 + (nt << 4) + lm] = acc[nt][r];
}

// ---------------- row LayerNorm (+optional fused RoPE), fp32 -> bf16 ----------------
__global__ void __launch_bounds__(256)
ln_rope_kernel(const float* __restrict__ src, bf16_t* __restrict__ dst, int W,
               const float* __restrict__ g, const float* __restrict__ bta, float eps,
               const float* __restrict__ cosb, const float* __restrict__ sinb) {
  __shared__ float red[256], red2[256];
  __shared__ float s_mu, s_rstd;
  const int row = blockIdx.x, t = threadIdx.x;
  const int chunk = W >> 8;
  const float* x = src + (long)row * W + t * chunk;

  float s = 0.f, s2 = 0.f;
  for (int i = 0; i < chunk; i++) { float v = x[i]; s += v; s2 += v * v; }
  red[t] = s; red2[t] = s2;
  __syncthreads();
  if (t == 0) {
    float a = 0.f, b2 = 0.f;
    for (int i = 0; i < 256; i++) { a += red[i]; b2 += red2[i]; }
    float mu = a / W;
    s_mu = mu;
    s_rstd = rsqrtf(b2 / W - mu * mu + eps);
  }
  __syncthreads();
  const float mu = s_mu, rstd = s_rstd;
  const int base = t * chunk;
  bf16_t* o = dst + (long)row * W + base;

  if (cosb) {
    const float* cp = cosb + (long)row * (HDD / 2);
    const float* sp = sinb + (long)row * (HDD / 2);
    for (int j = 0; j < chunk; j += 2) {
      int c0 = base + j;
      float x0 = (x[j]     - mu) * rstd * g[c0]     + bta[c0];
      float x1 = (x[j + 1] - mu) * rstd * g[c0 + 1] + bta[c0 + 1];
      int i2 = (c0 & (HDD - 1)) >> 1;
      float co = cp[i2], si = sp[i2];
      o[j]     = f2bf(x0 * co - x1 * si);
      o[j + 1] = f2bf(x0 * si + x1 * co);
    }
  } else {
    for (int j = 0; j < chunk; j++) {
      int c0 = base + j;
      o[j] = f2bf((x[j] - mu) * rstd * g[c0] + bta[c0]);
    }
  }
}

// ---------------- flash attention (bf16 in, fp32 out) ----------------
// Grid: (S/128, B*H). Block: 256 thr = 8 waves, each wave owns 16 query rows.
// kv_mod=0 -> kvh = h/NREP (jnp.repeat), kv_mod=1 -> kvh = h%KV (jnp.tile).
// K staged by async DMA (row-major, QK^T fragments read K rows contiguously);
// V staged transposed (PV fragments read d-rows of Vt contiguously).
__global__ void __launch_bounds__(256)
flash_attn_kernel(const bf16_t* __restrict__ Q, const bf16_t* __restrict__ Kb,
                  const bf16_t* __restrict__ Vb, const unsigned char* __restrict__ mask,
                  float* __restrict__ Out, int Lk, int kv_mod,
                  const float* __restrict__ gate, int accum) {
  __shared__ __attribute__((aligned(16))) bf16_t Ks[32][136];   // [key][d], padded
  __shared__ __attribute__((aligned(16))) bf16_t Vt[128][40];   // [d][key], padded
  __shared__ __attribute__((aligned(16))) bf16_t Ps[8][16][40]; // per-wave [q][key]
  __shared__ unsigned char smask[32];

  const int tid  = threadIdx.x;
  const int w    = tid >> 5, lane = tid & 31;
  const int half = lane >> 4, lm = lane & 15;
  const int h8 = half << 3;
  const int bh = blockIdx.y, bb = bh / HH, h = bh % HH;
  const int kvh = kv_mod ? (h % KVH) : (h / NREP);
  const int qbase = blockIdx.x * 128 + (w << 4);
  const float scale = 0.08838834764831845f; // 1/sqrt(128)

  // Resident Q A-fragments: 4 K-steps of 32 over HD=128 (global b128 pairs).
  v16bf qf[4];
  {
    const bf16_t* qrow = Q + (long)(bb * SS + qbase + lm) * (HH * HDD) + h * HDD;
#pragma unroll
    for (int ks = 0; ks < 4; ks++)
      qf[ks] = load_frag(qrow + ks * 32 + h8, qrow + ks * 32 + 16 + h8);
  }

  v8f zero = {0.f,0.f,0.f,0.f,0.f,0.f,0.f,0.f};
  v8f oacc[8];
#pragma unroll
  for (int i = 0; i < 8; i++) oacc[i] = zero;
  float mrow[8], lrow[8];
#pragma unroll
  for (int r = 0; r < 8; r++) { mrow[r] = -3.0e38f; lrow[r] = 0.f; }

  const int kr = tid >> 3, kc = (tid & 7) << 4;     // staging coords
  const long kvstride = (long)32 * (KVH * HDD);     // next 32-key block

  for (int kb0 = 0; kb0 < Lk; kb0 += 32) {
    long rowoff = (long)(bb * Lk + kb0 + kr) * (KVH * HDD) + kvh * HDD + kc;
    // stage K (32x128) via async DMA
    async_copy_32B(Kb + rowoff, &Ks[kr][kc]);
    { // stage V transposed
      bf16_t tmp[16];
      reinterpret_cast<uint4*>(tmp)[0] = reinterpret_cast<const uint4*>(Vb + rowoff)[0];
      reinterpret_cast<uint4*>(tmp)[1] = reinterpret_cast<const uint4*>(Vb + rowoff)[1];
#pragma unroll
      for (int i = 0; i < 16; i++) Vt[kc + i][kr] = tmp[i];
    }
    if (tid < 32) smask[tid] = mask[(long)bb * Lk + kb0 + tid];
    if (kb0 + 32 < Lk) { // prefetch next key block
      __builtin_prefetch(Kb + rowoff + kvstride, 0, 0);
      __builtin_prefetch(Vb + rowoff + kvstride, 0, 0);
    }
    wait_async_zero();
    __syncthreads();

    // S = Q K^T : 16 queries x 32 keys, two n-tiles, 4 WMMA each.
    v8f sc[2];
#pragma unroll
    for (int nt = 0; nt < 2; nt++) {
      sc[nt] = zero;
      const int kn = (nt << 4) + lm;
#pragma unroll
      for (int ks = 0; ks < 4; ks++) {
        v16bf kv = load_frag(&Ks[kn][ks * 32 + h8], &Ks[kn][ks * 32 + 16 + h8]);
        sc[nt] = __builtin_amdgcn_wmma_f32_16x16x32_bf16(
            false, qf[ks], false, kv, (short)0, sc[nt], false, false);
      }
    }

    const bool v0 = smask[lm] != 0;
    const bool v1 = smask[16 + lm] != 0;
    float p0a[8], p1a[8];
#pragma unroll
    for (int r = 0; r < 8; r++) {
      float s0 = v0 ? sc[0][r] * scale : -1.0e30f;
      float s1 = v1 ? sc[1][r] * scale : -1.0e30f;
      float pm = fmaxf(s0, s1);
#pragma unroll
      for (int off = 1; off < 16; off <<= 1) pm = fmaxf(pm, __shfl_xor(pm, off, 32));
      float mnew  = fmaxf(mrow[r], pm);
      float alpha = __expf(mrow[r] - mnew);
      float p0 = __expf(s0 - mnew), p1 = __expf(s1 - mnew);
      float ps = p0 + p1;
#pragma unroll
      for (int off = 1; off < 16; off <<= 1) ps += __shfl_xor(ps, off, 32);
      lrow[r] = lrow[r] * alpha + ps;
      mrow[r] = mnew;
#pragma unroll
      for (int nt = 0; nt < 8; nt++) oacc[nt][r] *= alpha;
      p0a[r] = p0; p1a[r] = p1;
    }

    // Transpose P (C-layout -> A-layout) through per-wave LDS region.
#pragma unroll
    for (int r = 0; r < 8; r++) {
      Ps[w][r + (half << 3)][lm]      = f2bf(p0a[r]);
      Ps[w][r + (half << 3)][16 + lm] = f2bf(p1a[r]);
    }
    v16bf pf = load_frag(&Ps[w][lm][h8], &Ps[w][lm][16 + h8]);

    // O += P * V : 16x32 times 32x128 -> 8 WMMA, B-fragments from Vt rows.
#pragma unroll
    for (int nt = 0; nt < 8; nt++) {
      const int vn = (nt << 4) + lm;
      v16bf vv = load_frag(&Vt[vn][h8], &Vt[vn][16 + h8]);
      oacc[nt] = __builtin_amdgcn_wmma_f32_16x16x32_bf16(
          false, pf, false, vv, (short)0, oacc[nt], false, false);
    }
    __syncthreads();
  }

  const float gf = gate ? tanhf(gate[h]) : 1.0f;
#pragma unroll
  for (int nt = 0; nt < 8; nt++)
#pragma unroll
    for (int r = 0; r < 8; r++) {
      long row = (long)bb * SS + qbase + r + (half << 3);
      long idx = row * (HH * HDD) + h * HDD + (nt << 4) + lm;
      float val = oacc[nt][r] / fmaxf(lrow[r], 1e-20f) * gf;
      if (accum) Out[idx] += val; else Out[idx] = val;
    }
}

// ---------------- host-side orchestration ----------------
extern "C" void kernel_launch(void* const* d_in, const int* in_sizes, int n_in,
                              void* d_out, int out_size, void* d_ws, size_t ws_size,
                              hipStream_t stream) {
  const float* x        = (const float*)d_in[0];
  const unsigned char* x_mask = (const unsigned char*)d_in[1];
  const float* fcos     = (const float*)d_in[2];
  const float* fsin     = (const float*)d_in[3];
  const float* y        = (const float*)d_in[4];
  const unsigned char* y_mask = (const unsigned char*)d_in[5];
  const float* wq       = (const float*)d_in[6];
  const float* wk       = (const float*)d_in[7];
  const float* wv       = (const float*)d_in[8];
  const float* wk_y     = (const float*)d_in[9];
  const float* wv_y     = (const float*)d_in[10];
  const float* wo       = (const float*)d_in[11];
  const float* gate     = (const float*)d_in[12];
  const float* q_g      = (const float*)d_in[13];
  const float* q_b      = (const float*)d_in[14];
  const float* k_g      = (const float*)d_in[15];
  const float* k_b      = (const float*)d_in[16];
  const float* ky_g     = (const float*)d_in[17];
  const float* ky_b     = (const float*)d_in[18];
  float* out            = (float*)d_out;

  const long ROWS  = (long)BB * SS;      // 4096
  const long YROWS = (long)BB * YLL;     // 512

  char* p = (char*)d_ws;
  auto alloc = [&](size_t bytes) -> char* {
    char* r = p; p += (bytes + 255) & ~(size_t)255; return r;
  };
  bf16_t* xb    = (bf16_t*)alloc(ROWS * DD * 2);
  bf16_t* yb    = (bf16_t*)alloc(YROWS * YDD * 2);
  bf16_t* wqb   = (bf16_t*)alloc((size_t)DD * DD * 2);
  bf16_t* wkb   = (bf16_t*)alloc((size_t)DD * (KVH * HDD) * 2);
  bf16_t* wvb   = (bf16_t*)alloc((size_t)DD * (KVH * HDD) * 2);
  bf16_t* wkyb  = (bf16_t*)alloc((size_t)YDD * (KVH * HDD) * 2);
  bf16_t* wvyb  = (bf16_t*)alloc((size_t)YDD * (KVH * HDD) * 2);
  bf16_t* wob   = (bf16_t*)alloc((size_t)DD * DD * 2);
  float*  qf    = (float*)alloc(ROWS * DD * 4);
  float*  kf    = (float*)alloc(ROWS * (KVH * HDD) * 4);
  float*  vf    = (float*)alloc(ROWS * (KVH * HDD) * 4);
  float*  ykf   = (float*)alloc(YROWS * (KVH * HDD) * 4);
  float*  yvf   = (float*)alloc(YROWS * (KVH * HDD) * 4);
  bf16_t* qb    = (bf16_t*)alloc(ROWS * DD * 2);
  bf16_t* kb    = (bf16_t*)alloc(ROWS * (KVH * HDD) * 2);
  bf16_t* vb    = (bf16_t*)alloc(ROWS * (KVH * HDD) * 2);
  bf16_t* ykb   = (bf16_t*)alloc(YROWS * (KVH * HDD) * 2);
  bf16_t* yvb   = (bf16_t*)alloc(YROWS * (KVH * HDD) * 2);
  float*  attnf = (float*)alloc(ROWS * DD * 4);
  bf16_t* attnb = (bf16_t*)alloc(ROWS * DD * 2);

  auto cvt = [&](const float* s, bf16_t* d, long n) {
    int blocks = (int)((n + 1023) / 1024);
    if (blocks > 4096) blocks = 4096;
    cvt_bf16_kernel<<<blocks, 256, 0, stream>>>(s, d, n);
  };
  auto gemm = [&](const bf16_t* A, const bf16_t* Bm, float* C, int M, int N, int K) {
    gemm_bf16_kernel<<<dim3(N / 64, M / 128), 256, 0, stream>>>(A, Bm, C, M, N, K);
  };

  // 1. fp32 -> bf16 for activations and weights
  cvt(x,    xb,   ROWS * DD);
  cvt(y,    yb,   YROWS * YDD);
  cvt(wq,   wqb,  (long)DD * DD);
  cvt(wk,   wkb,  (long)DD * KVH * HDD);
  cvt(wv,   wvb,  (long)DD * KVH * HDD);
  cvt(wk_y, wkyb, (long)YDD * KVH * HDD);
  cvt(wv_y, wvyb, (long)YDD * KVH * HDD);
  cvt(wo,   wob,  (long)DD * DD);

  // 2. projections (WMMA GEMMs)
  gemm(xb, wqb,  qf,  (int)ROWS,  HH * HDD,  DD);
  gemm(xb, wkb,  kf,  (int)ROWS,  KVH * HDD, DD);
  gemm(xb, wvb,  vf,  (int)ROWS,  KVH * HDD, DD);
  gemm(yb, wkyb, ykf, (int)YROWS, KVH * HDD, YDD);
  gemm(yb, wvyb, yvf, (int)YROWS, KVH * HDD, YDD);

  // 3. LayerNorm (+RoPE for q/k), convert to bf16
  ln_rope_kernel<<<(int)ROWS,  256, 0, stream>>>(qf,  qb,  HH * HDD,  q_g,  q_b,  1e-5f, fcos, fsin);
  ln_rope_kernel<<<(int)ROWS,  256, 0, stream>>>(kf,  kb,  KVH * HDD, k_g,  k_b,  1e-5f, fcos, fsin);
  ln_rope_kernel<<<(int)YROWS, 256, 0, stream>>>(ykf, ykb, KVH * HDD, ky_g, ky_b, 1e-6f, nullptr, nullptr);
  cvt(vf,  vb,  ROWS * KVH * HDD);
  cvt(yvf, yvb, YROWS * KVH * HDD);

  // 4. flash attention: self (repeat mapping), then gated cross (tile mapping), accumulated
  dim3 agrid(SS / 128, BB * HH);
  flash_attn_kernel<<<agrid, 256, 0, stream>>>(qb, kb,  vb,  x_mask, attnf, SS,  0, nullptr, 0);
  flash_attn_kernel<<<agrid, 256, 0, stream>>>(qb, ykb, yvb, y_mask, attnf, YLL, 1, gate,    1);

  // 5. output projection
  cvt(attnf, attnb, ROWS * DD);
  gemm(attnb, wob, out, (int)ROWS, DD, DD);
}
